// Variable_Attention_35983236006188
// MI455X (gfx1250) — matrix operand order repaired and checked by reading, not compiled, not run on hardware
//
#include <hip/hip_runtime.h>
#include <stdint.h>

// Problem constants (match reference)
#define B_  32
#define N_  1024
#define D_  256
#define H_  8
#define E_  256
#define HE_ 2048   // H*E

// Scratch layout (float offsets)
#define OFF_XBAR   0                       // [B, D]            8192
#define OFF_S      8192                    // scalar
#define OFF_KBAR   8448                    // [B, HE]           65536
#define OFF_CQ     73984                   // [B, H]            256
#define OFF_WQBAR  74240                   // [B, H, D]         65536
#define OFF_LOGITS 139776                  // [B, H, N] (beta in-place) 262144
#define OFF_XV     401920                  // [B, H, D]         65536
#define OFF_RESULT 467456                  // [B, HE]           65536
// total 532992 floats ~ 2.1 MB (L2-resident)

typedef __attribute__((ext_vector_type(16))) _Float16 v16h;
typedef __attribute__((ext_vector_type(8)))  float    v8f;
typedef unsigned int uint32x4 __attribute__((ext_vector_type(4)));
typedef int          int32x8  __attribute__((ext_vector_type(8)));
typedef int          int32x4  __attribute__((ext_vector_type(4)));

// ---- WMMA fragment helpers (ISA 7.12.2 layouts, wave32) -------------------
// A (16x32 f16): lane L holds row M=L&15; VGPR v holds K pair:
//   v<4 : K = 2v + 8g ; v>=4 : K = 16 + 2(v-4) + 8g   (g = L>>4)
__device__ __forceinline__ v16h fragA(const float* p, int g, float mask) {
  v16h r;
#pragma unroll
  for (int v = 0; v < 8; ++v) {
    int k = (v < 4) ? (2 * v + 8 * g) : (16 + 2 * (v - 4) + 8 * g);
    r[2 * v]     = (_Float16)(mask * p[k]);
    r[2 * v + 1] = (_Float16)(mask * p[k + 1]);
  }
  return r;
}
// B (32x16 f16): lane L holds col N=L&15; VGPR v holds K pair same as A (v<4: 2v+8g; v>=4: 16+2(v-4)+8g).
__device__ __forceinline__ v16h fragB(const float* p, long strideK, int g) {
  v16h r;
#pragma unroll
  for (int v = 0; v < 8; ++v) {
    long k = (v < 4) ? (2 * v + 8 * g) : (16 + 2 * (v - 4) + 8 * g);
    r[2 * v]     = (_Float16)p[k * strideK];
    r[2 * v + 1] = (_Float16)p[(k + 1) * strideK];
  }
  return r;
}
__device__ __forceinline__ v8f wmma16(v16h a, v16h b, v8f c) {
  return __builtin_amdgcn_wmma_f32_16x16x32_f16(false, a, false, b, (short)0, c,
                                                false, false);
}

// ---- TDM: async 2D f32 tile load Global -> LDS (D# per ISA 08 §8.3/8.4) ---
// pad_bits: OR'd into group1 word0 (pad_enable/pad_interval/pad_amount).
__device__ __forceinline__ void tdm_load_2d(uint32_t lds_byte_off, const float* gsrc,
                                            uint32_t tensor_w, uint32_t tensor_h,
                                            uint32_t tile_w, uint32_t tile_h,
                                            uint32_t row_stride, uint32_t pad_bits) {
  uint64_t ga = (uint64_t)(uintptr_t)gsrc;
  uint32x4 g0;
  g0[0] = 1u;                                               // count=1, user mode
  g0[1] = lds_byte_off;                                     // lds_addr (bytes)
  g0[2] = (uint32_t)(ga & 0xFFFFFFFFu);                     // global_addr[31:0]
  g0[3] = (uint32_t)((ga >> 32) & 0x01FFFFFFu) | (2u << 30);// addr[56:32] | type=2
  int32x8 g1;
  g1[0] = (int)((2u << 16) | pad_bits);                     // data_size=4B, wg_mask=0
  g1[1] = (int)((tensor_w & 0xFFFFu) << 16);                // tensor_dim0[15:0]
  g1[2] = (int)(((tensor_w >> 16) & 0xFFFFu) |              // tensor_dim0[31:16]
                ((tensor_h & 0xFFFFu) << 16));              // tensor_dim1[15:0]
  g1[3] = (int)(((tensor_h >> 16) & 0xFFFFu) |              // tensor_dim1[31:16]
                ((tile_w & 0xFFFFu) << 16));                // tile_dim0
  g1[4] = (int)(tile_h & 0xFFFFu);                          // tile_dim1 (tile_dim2=0)
  g1[5] = (int)row_stride;                                  // tensor_dim0_stride[31:0]
  g1[6] = 0;                                                // stride hi / dim1_stride
  g1[7] = 0;
  int32x4 gz4 = {0, 0, 0, 0};                               // 2D: groups 2/3 unused
  int32x8 gz8 = {0, 0, 0, 0, 0, 0, 0, 0};
  __builtin_amdgcn_tensor_load_to_lds(g0, g1, gz4, gz4, gz8, 0);
}

// ---- k_init: S = sum(ws) --------------------------------------------------
__global__ void k_init(const float* __restrict__ wsum, float* __restrict__ scr) {
  __shared__ float red[1024];
  int t = threadIdx.x;
  red[t] = wsum[t];                 // N_ == 1024 == blockDim
  __syncthreads();
  for (int s = 512; s > 0; s >>= 1) {
    if (t < s) red[t] += red[t + s];
    __syncthreads();
  }
  if (t == 0) scr[OFF_S] = red[0];
}

// ---- k_xbar: xbar[b,d] = sum_n ws[n] * x[b,n,d]  (pass 1 over x) ----------
__global__ void k_xbar(const float* __restrict__ x, const float* __restrict__ wsum,
                       float* __restrict__ scr) {
  int b = blockIdx.x, d = threadIdx.x;
  const float* xb = x + (size_t)b * N_ * D_ + d;
  float acc = 0.f;
  for (int n = 0; n < N_; ++n) acc = fmaf(wsum[n], xb[(size_t)n * D_], acc);
  scr[OFF_XBAR + b * D_ + d] = acc;
}

// ---- k_kbar: kbar[b,he] = xbar[b,:]·Wk[he,:] + S*bk[he]   (GEMM 32x2048x256)
__global__ void k_kbar(const float* __restrict__ Wk, const float* __restrict__ bk,
                       float* __restrict__ scr) {
  int lane = threadIdx.x, g = lane >> 4, l = lane & 15;
  int n0 = blockIdx.x * 16, m0 = blockIdx.y * 16;
  float S = scr[OFF_S];
  const float* arow = scr + OFF_XBAR + (size_t)(m0 + l) * D_;   // contiguous K
  const float* bcol = Wk + (size_t)(n0 + l) * D_;               // contiguous K
  v8f c = {};
  for (int k0 = 0; k0 < D_; k0 += 32)
    c = wmma16(fragA(arow + k0, g, 1.f), fragB(bcol + k0, 1, g), c);
  int he = n0 + l;
  float sb = S * bk[he];
#pragma unroll
  for (int r = 0; r < 8; ++r) {
    int b = m0 + r + 8 * g;
    scr[OFF_KBAR + (size_t)b * HE_ + he] = c[r] + sb;
  }
}

// ---- k_cq: cq[b,h] = bq[hE:..]·kbar[b,h,:] --------------------------------
__global__ void k_cq(const float* __restrict__ bq, float* __restrict__ scr) {
  int t = threadIdx.x, b = t >> 3, h = t & 7;
  const float* kb = scr + OFF_KBAR + (size_t)b * HE_ + h * E_;
  float acc = 0.f;
  for (int e = 0; e < E_; ++e) acc = fmaf(bq[h * E_ + e], kb[e], acc);
  scr[OFF_CQ + b * H_ + h] = acc;
}

// ---- k_wqbar: wqbar[b,h,d] = sum_e Wq[hE+e,d]*kbar[b,h,e] -----------------
__global__ void k_wqbar(const float* __restrict__ Wq, float* __restrict__ scr) {
  int lane = threadIdx.x, g = lane >> 4, l = lane & 15;
  int h = blockIdx.x, d0 = blockIdx.y * 16, m0 = blockIdx.z * 16;
  const float* arow = scr + OFF_KBAR + (size_t)(m0 + l) * HE_ + h * E_; // contiguous K
  const float* bcol = Wq + (size_t)h * E_ * D_ + d0 + l;               // strideK = D_
  v8f c = {};
  for (int k0 = 0; k0 < E_; k0 += 32)
    c = wmma16(fragA(arow + k0, g, 1.f), fragB(bcol + (size_t)k0 * D_, D_, g), c);
#pragma unroll
  for (int r = 0; r < 8; ++r) {
    int b = m0 + r + 8 * g;
    scr[OFF_WQBAR + ((size_t)b * H_ + h) * D_ + d0 + l] = c[r];
  }
}

// ---- k_logits: logits[b,h,n] = scale*(x[b,n,:]·wqbar[b,h,:] + cq) + bs ----
// Pass 2 over x. The 16x256 f32 x-tile is staged to LDS by the Tensor Data
// Mover with 1-DWORD padding every 256 DWORDs (row stride 257 floats) so the
// A-fragment row reads are bank-conflict-free.
#define LROWS 257
__global__ void k_logits(const float* __restrict__ x, const float* __restrict__ bs,
                         float* __restrict__ scr) {
  __shared__ float tile[16 * LROWS];                                   // 16448 B
  int lane = threadIdx.x, g = lane >> 4, l = lane & 15;
  int b = blockIdx.x, n0 = blockIdx.y * 16;
  // TDM: tile = x[b, n0:n0+16, 0:256]; pad_enable, interval=256 DW, amount=1 DW
  uint32_t pad = (1u << 20) | (7u << 22) | (0u << 25);
  tdm_load_2d((uint32_t)(uintptr_t)(void*)tile,
              x + ((size_t)b * N_ + n0) * D_,
              D_, N_, D_, 16, D_, pad);
  int hh = l & 7;                                                      // cols 8..15 dup
  const float* bcol = scr + OFF_WQBAR + ((size_t)b * H_ + hh) * D_;    // contiguous K
  __builtin_amdgcn_s_wait_tensorcnt(0);
  const float* arow = tile + l * LROWS;                                // LDS row
  v8f c = {};
  for (int k0 = 0; k0 < D_; k0 += 32)
    c = wmma16(fragA(arow + k0, g, 1.f), fragB(bcol + k0, 1, g), c);
  if (l < H_) {
    float add = 0.0625f /* 1/sqrt(256) */ * scr[OFF_CQ + b * H_ + l];  // scale*cq
    float bias = bs[0];
#pragma unroll
    for (int r = 0; r < 8; ++r) {
      int n = n0 + r + 8 * g;
      scr[OFF_LOGITS + ((size_t)b * H_ + l) * N_ + n] = 0.0625f * c[r] + add + bias;
    }
  }
}

// ---- k_softmax over n (1024) per (b,h); beta written in place -------------
__global__ void k_softmax(float* __restrict__ scr) {
  __shared__ float red[256];
  int t = threadIdx.x;
  float* lg = scr + OFF_LOGITS + (size_t)blockIdx.x * N_;
  float m = -1e30f;
  for (int i = t; i < N_; i += 256) m = fmaxf(m, lg[i]);
  red[t] = m; __syncthreads();
  for (int s = 128; s > 0; s >>= 1) { if (t < s) red[t] = fmaxf(red[t], red[t + s]); __syncthreads(); }
  m = red[0]; __syncthreads();
  float sum = 0.f;
  for (int i = t; i < N_; i += 256) { float e = __expf(lg[i] - m); lg[i] = e; sum += e; }
  red[t] = sum; __syncthreads();
  for (int s = 128; s > 0; s >>= 1) { if (t < s) red[t] += red[t + s]; __syncthreads(); }
  float inv = 1.f / red[0];
  for (int i = t; i < N_; i += 256) lg[i] *= inv;
}

// ---- k_xv: xv[b,h,d] = sum_n beta[b,h,n]*x[b,n,d]  (pass 3 over x) --------
// A = beta [8 rows padded to 16, K=1024]; B = x column block [K=1024, 16 d].
// x chunks (128 rows x 16 cols, 8 KB) are double-buffered through LDS via TDM:
// prefetch chunk i+1, s_wait_tensorcnt(1), WMMA on chunk i.
#define XCH 128
__global__ void k_xv(const float* __restrict__ x, float* __restrict__ scr) {
  __shared__ float tile[2 * XCH * 16];                                  // 16 KB
  int lane = threadIdx.x, g = lane >> 4, l = lane & 15;
  int b = blockIdx.x, d0 = blockIdx.y * 16;
  float amask = (l < H_) ? 1.f : 0.f;                                   // zero pad rows
  const float* arow = scr + OFF_LOGITS + ((size_t)b * H_ + (l & 7)) * N_;
  const float* xbase = x + (size_t)b * N_ * D_ + d0;
  uint32_t lds0 = (uint32_t)(uintptr_t)(void*)tile;
  // prologue: fetch chunk 0
  tdm_load_2d(lds0, xbase, D_, N_, 16, XCH, D_, 0);
  int cur = 0;
  v8f c = {};
  for (int ck = 0; ck < N_; ck += XCH) {
    if (ck + XCH < N_) {
      tdm_load_2d(lds0 + (uint32_t)((cur ^ 1) * XCH * 16 * 4),
                  xbase + (size_t)(ck + XCH) * D_, D_, N_, 16, XCH, D_, 0);
      __builtin_amdgcn_s_wait_tensorcnt(1);                             // chunk ck done
    } else {
      __builtin_amdgcn_s_wait_tensorcnt(0);
    }
    const float* bt = tile + cur * XCH * 16;
    for (int k0 = 0; k0 < XCH; k0 += 32)
      c = wmma16(fragA(arow + ck + k0, g, amask), fragB(bt + k0 * 16 + l, 16, g), c);
    cur ^= 1;
  }
  if (g == 0) {                                                         // rows 0..7 = h
#pragma unroll
    for (int r = 0; r < 8; ++r)
      scr[OFF_XV + ((size_t)b * H_ + r) * D_ + d0 + l] = c[r];
  }
}

// ---- k_result: result[b,hE+e] = sum_d Wv[hE+e,d]*xv[b,h,d] + bv -----------
__global__ void k_result(const float* __restrict__ Wv, const float* __restrict__ bv,
                         float* __restrict__ scr) {
  int lane = threadIdx.x, g = lane >> 4, l = lane & 15;
  int h = blockIdx.x, e0 = blockIdx.y * 16, m0 = blockIdx.z * 16;
  const float* arow = scr + OFF_XV + ((size_t)(m0 + l) * H_ + h) * D_;  // contiguous K
  const float* bcol = Wv + ((size_t)(h * E_ + e0 + l)) * D_;            // contiguous K
  v8f c = {};
  for (int k0 = 0; k0 < D_; k0 += 32)
    c = wmma16(fragA(arow + k0, g, 1.f), fragB(bcol + k0, 1, g), c);
  int he = h * E_ + e0 + l;
  float bias = bv[he];
#pragma unroll
  for (int r = 0; r < 8; ++r) {
    int b = m0 + r + 8 * g;
    scr[OFF_RESULT + (size_t)b * HE_ + he] = c[r] + bias;
  }
}

// ---- k_out: out[b,d] = sum_he result[b,he]*Wo[d,he] + bo[d] ---------------
__global__ void k_out(const float* __restrict__ Wo, const float* __restrict__ bo,
                      const float* __restrict__ scr, float* __restrict__ out) {
  int lane = threadIdx.x, g = lane >> 4, l = lane & 15;
  int d0 = blockIdx.x * 16, m0 = blockIdx.y * 16;
  const float* arow = scr + OFF_RESULT + (size_t)(m0 + l) * HE_;        // contiguous K
  const float* bcol = Wo + (size_t)(d0 + l) * HE_;                      // contiguous K
  v8f c = {};
  for (int k0 = 0; k0 < HE_; k0 += 32)
    c = wmma16(fragA(arow + k0, g, 1.f), fragB(bcol + k0, 1, g), c);
  int d = d0 + l;
  float bias = bo[d];
#pragma unroll
  for (int r = 0; r < 8; ++r) {
    int b = m0 + r + 8 * g;
    out[(size_t)b * D_ + d] = c[r] + bias;
  }
}

extern "C" void kernel_launch(void* const* d_in, const int* in_sizes, int n_in,
                              void* d_out, int out_size, void* d_ws, size_t ws_size,
                              hipStream_t stream) {
  const float* x   = (const float*)d_in[0];
  const float* Wq  = (const float*)d_in[1];
  const float* bq  = (const float*)d_in[2];
  const float* Wk  = (const float*)d_in[3];
  const float* bk  = (const float*)d_in[4];
  const float* Wv  = (const float*)d_in[5];
  const float* bv  = (const float*)d_in[6];
  const float* wsm = (const float*)d_in[7];
  const float* bs  = (const float*)d_in[8];
  const float* Wo  = (const float*)d_in[9];
  const float* bo  = (const float*)d_in[10];
  float* out = (float*)d_out;
  float* scr = (float*)d_ws;

  k_init   <<<dim3(1),              dim3(1024), 0, stream>>>(wsm, scr);
  k_xbar   <<<dim3(B_),             dim3(256),  0, stream>>>(x, wsm, scr);
  k_kbar   <<<dim3(HE_/16, 2),      dim3(32),   0, stream>>>(Wk, bk, scr);
  k_cq     <<<dim3(1),              dim3(256),  0, stream>>>(bq, scr);
  k_wqbar  <<<dim3(H_, D_/16, 2),   dim3(32),   0, stream>>>(Wq, scr);
  k_logits <<<dim3(B_, N_/16),      dim3(32),   0, stream>>>(x, bs, scr);
  k_softmax<<<dim3(B_*H_),          dim3(256),  0, stream>>>(scr);
  k_xv     <<<dim3(B_, D_/16),      dim3(32),   0, stream>>>(x, scr);
  k_result <<<dim3(H_, E_/16, 2),   dim3(32),   0, stream>>>(Wv, bv, scr);
  k_out    <<<dim3(D_/16, 2),       dim3(32),   0, stream>>>(Wo, bo, scr, out);
}
